// SimplePaiNNModel_37220186587476
// MI455X (gfx1250) — matrix operand order repaired and checked by reading, not compiled, not run on hardware
//
#include <hip/hip_runtime.h>
#include <hip/hip_bf16.h>
#include <stdint.h>

// Problem constants (must match reference)
#define B_   16
#define N_   512
#define D_   512
#define L_   4
#define E_   131072
#define PAD_ 127
#define BN_  (B_ * N_)   // 8192

typedef __attribute__((ext_vector_type(16))) __bf16          v16bf;
typedef __attribute__((ext_vector_type(16))) unsigned short  v16us;
typedef __attribute__((ext_vector_type(8)))  float           v8f;
typedef int v4i_gcc __attribute__((vector_size(16)));   // matches builtin proto

#define GLOBAL_AS __attribute__((address_space(1)))
#define LDS_AS    __attribute__((address_space(3)))

// ---- gfx1250 async global->LDS staging (guarded; falls back to sync copy) ----
#if defined(__gfx1250__) && __has_builtin(__builtin_amdgcn_global_load_async_to_lds_b128)
#define HAVE_ASYNC_LDS 1
#else
#define HAVE_ASYNC_LDS 0
#endif

#if HAVE_ASYNC_LDS && __has_builtin(__builtin_amdgcn_s_wait_asynccnt)
#define ASYNC_WAIT() __builtin_amdgcn_s_wait_asynccnt(0)
#elif HAVE_ASYNC_LDS
#define ASYNC_WAIT() asm volatile("s_wait_asynccnt 0" ::: "memory")
#else
#define ASYNC_WAIT() ((void)0)
#endif

__device__ __forceinline__ void copy16_to_lds(const unsigned short* g,
                                              unsigned short* l) {
#if HAVE_ASYNC_LDS
  // Per-lane 16-byte async copy, tracked on ASYNCcnt.
  // Builtin signature (from compiler diagnostic): param0 = v4i __device__*
  // (global), param1 = v4i LDS pointer, then imm offset + imm cpol.
  // Generic shared-memory address: low 32 bits are the wave-relative LDS
  // offset (flat LDS aperture rule), so the integer route yields a valid
  // addrspace(3) pointer value.
  __builtin_amdgcn_global_load_async_to_lds_b128(
      (GLOBAL_AS v4i_gcc*)(uintptr_t)g,
      (LDS_AS v4i_gcc*)(uintptr_t)(unsigned)(uintptr_t)l,
      0, 0);
#else
  *(uint4*)l = *(const uint4*)g;
#endif
}

__device__ __forceinline__ unsigned short f2bf(float f) {
  unsigned int u = __float_as_uint(f);
  u += 0x7FFFu + ((u >> 16) & 1u);   // round-to-nearest-even
  return (unsigned short)(u >> 16);
}

// ---------------------------------------------------------------------------
// fp32 [L,K,Nc] -> bf16 [L,Nc,K]  (convert + transpose, once per launch).
// Pre-transposing the weights makes every GEMM tile-staging transfer and
// every B-fragment LDS read fully contiguous.
// ---------------------------------------------------------------------------
__global__ void f2bf_transpose_kernel(const float* __restrict__ src,
                                      unsigned short* __restrict__ dst,
                                      int K, int Nc) {
  int i = blockIdx.x * blockDim.x + threadIdx.x;   // over L*K*Nc
  int l = i / (K * Nc);
  int r = i - l * K * Nc;
  int k = r / Nc;
  int n = r - k * Nc;
  dst[(size_t)l * Nc * K + (size_t)n * K + k] = f2bf(src[i]);
}

// ---------------------------------------------------------------------------
// Embedding gather: x[bn,d] = embed[tok[bn], d]; also bf16 copy of x.
// ---------------------------------------------------------------------------
__global__ void embed_kernel(const int* __restrict__ tok,
                             const float* __restrict__ emb,
                             float* __restrict__ x,
                             unsigned short* __restrict__ xbf) {
  int i  = blockIdx.x * blockDim.x + threadIdx.x;     // over BN*D/4
  int bn = i / (D_ / 4);
  int dc = (i % (D_ / 4)) * 4;
  int tk = tok[bn];
  float4 e = *(const float4*)(emb + (size_t)tk * D_ + dc);
  *(float4*)(x + (size_t)bn * D_ + dc) = e;
  unsigned short h0 = f2bf(e.x), h1 = f2bf(e.y), h2 = f2bf(e.z), h3 = f2bf(e.w);
  uint2 packed;
  packed.x = (unsigned)h0 | ((unsigned)h1 << 16);
  packed.y = (unsigned)h2 | ((unsigned)h3 << 16);
  *(uint2*)(xbf + (size_t)bn * D_ + dc) = packed;
}

// ---------------------------------------------------------------------------
// v[bn,d] = coord[bn,:] @ vec_proj_w[:,d] + vec_proj_b[d]
// ---------------------------------------------------------------------------
__global__ void vecproj_kernel(const float* __restrict__ coord,
                               const float* __restrict__ vw,
                               const float* __restrict__ vb,
                               float* __restrict__ v) {
  int i  = blockIdx.x * blockDim.x + threadIdx.x;     // over BN*D
  int bn = i / D_;
  int d  = i - bn * D_;
  float c0 = coord[bn * 3 + 0];
  float c1 = coord[bn * 3 + 1];
  float c2 = coord[bn * 3 + 2];
  v[i] = c0 * vw[d] + c1 * vw[D_ + d] + c2 * vw[2 * D_ + d] + vb[d];
}

// ---------------------------------------------------------------------------
// BF16 WMMA GEMM: Dout[M,Nc] = A[M,K](bf16) * Bt[Nc,K](bf16, pre-transposed)
//                              + bias[Nc] (+ Cadd)
// Block: 256 threads (8 waves). Tile 128(M) x 64(N) x 32(K).
// Each wave: 32x32 patch = 2x2 v_wmma_f32_16x16x32_bf16.
// Staging: per-lane 16B async copies to LDS (ASYNCcnt) when available.
// ---------------------------------------------------------------------------
#define TM 128
#define TN 64
#define TK 32
#define APAD 40   // LDS row stride (ushorts): 80 B, 16B-aligned, padded

union FragU { uint4 u[2]; v16us v; };

__global__ __launch_bounds__(256)
void gemm_bf16_kernel(const unsigned short* __restrict__ A,   // M x K
                      const unsigned short* __restrict__ Bt,  // Nc x K
                      const float* __restrict__ bias,         // Nc
                      const float* __restrict__ Cadd,         // M x Nc or null
                      float* __restrict__ Dout,               // M x Nc
                      int M, int Nc, int K) {
  __shared__ __align__(16) unsigned short As[TM][APAD];
  __shared__ __align__(16) unsigned short Bs[TN][APAD];   // [n][k]

  const int m0   = blockIdx.y * TM;
  const int n0   = blockIdx.x * TN;
  const int t    = threadIdx.x;
  const int lane = t & 31;
  const int wave = t >> 5;
  const int wm   = wave & 3;     // 0..3  (M sub-tile, 32 rows each)
  const int wn   = wave >> 2;    // 0..1  (N sub-tile, 32 cols each)
  const int lmod = lane & 15;
  const int lhi  = lane >> 4;    // 0 or 1

  // Staging assignments (contiguous 16B units)
  const int arow = t >> 1, ahalf = t & 1;   // A: 128 rows x 2 halves (16 elems)
  const int brow = t >> 2, bseg = t & 3;    // B: 64 rows x 4 segments (8 elems)

  v8f acc[2][2] = {};

  for (int k0 = 0; k0 < K; k0 += TK) {
    // ---- stage A tile (128 x 32): two 16B copies per thread ----
    const unsigned short* ga = A + (size_t)(m0 + arow) * K + k0 + ahalf * 16;
    copy16_to_lds(ga,     &As[arow][ahalf * 16]);
    copy16_to_lds(ga + 8, &As[arow][ahalf * 16 + 8]);
    // ---- stage B tile (64 x 32): one 16B copy per thread ----
    const unsigned short* gb = Bt + (size_t)(n0 + brow) * K + k0 + bseg * 8;
    copy16_to_lds(gb, &Bs[brow][bseg * 8]);

    if (k0 + TK < K) {
      __builtin_prefetch(ga + TK, 0, 3);
      __builtin_prefetch(gb + TK, 0, 3);
    }

    ASYNC_WAIT();
    __syncthreads();

    // ---- build fragments (per ISA per-lane layouts) and multiply ----
    FragU afr[2], bfr[2];
#pragma unroll
    for (int mi = 0; mi < 2; ++mi) {
      const int r    = wm * 32 + mi * 16 + lmod;
      const int off0 = lhi * 8;     // A: K runs [off0,off0+7],[off0+16,off0+23]
      afr[mi].u[0] = *(const uint4*)&As[r][off0];
      afr[mi].u[1] = *(const uint4*)&As[r][off0 + 16];
    }
#pragma unroll
    for (int ni = 0; ni < 2; ++ni) {
      const int n   = wn * 32 + ni * 16 + lmod;
      const int off = lhi * 8;      // B: K runs [off,off+7],[off+16,off+23]
      bfr[ni].u[0] = *(const uint4*)&Bs[n][off];
      bfr[ni].u[1] = *(const uint4*)&Bs[n][off + 16];
    }
#pragma unroll
    for (int mi = 0; mi < 2; ++mi)
#pragma unroll
      for (int ni = 0; ni < 2; ++ni)
        acc[mi][ni] = __builtin_amdgcn_wmma_f32_16x16x32_bf16(
            false, __builtin_bit_cast(v16bf, afr[mi].v),
            false, __builtin_bit_cast(v16bf, bfr[ni].v),
            (short)0, acc[mi][ni], false, false);
    __syncthreads();
  }

  // ---- epilogue: C/D layout -> global.  m = vgpr_j + 8*lhi ----
#pragma unroll
  for (int mi = 0; mi < 2; ++mi)
#pragma unroll
    for (int ni = 0; ni < 2; ++ni) {
      const int gn = n0 + wn * 32 + ni * 16 + lmod;
      const float bb = bias[gn];
#pragma unroll
      for (int j = 0; j < 8; ++j) {
        const int gm = m0 + wm * 32 + mi * 16 + j + lhi * 8;
        float val = acc[mi][ni][j] + bb;
        size_t off = (size_t)gm * Nc + gn;
        if (Cadd) val += Cadd[off];
        Dout[off] = val;
      }
    }
}

// ---------------------------------------------------------------------------
// Edge phase: one 128-thread block per edge (4 floats of D per thread).
// ew = dot(q[row], k[col]);  agg_s[row] += ew*v_s[col];
// agg_v[row] += ew*v_v[col]*v[col]
// ---------------------------------------------------------------------------
__global__ __launch_bounds__(128)
void edge_kernel(const int* __restrict__ edge_index,
                 const float* __restrict__ q,    // BN x D
                 const float* __restrict__ kv,   // BN x 3D  (k | v_s | v_v)
                 const float* __restrict__ v,    // BN x D
                 float* __restrict__ agg_s,
                 float* __restrict__ agg_v) {
  const int e   = blockIdx.x;
  const int row = edge_index[e];
  const int col = edge_index[E_ + e];
  const int t   = threadIdx.x;

  const float* kvc = kv + (size_t)col * 3 * D_;
  float4 q4 = ((const float4*)(q + (size_t)row * D_))[t];
  float4 k4 = ((const float4*)kvc)[t];
  float part = q4.x * k4.x + q4.y * k4.y + q4.z * k4.z + q4.w * k4.w;
#pragma unroll
  for (int off = 16; off > 0; off >>= 1) part += __shfl_xor(part, off, 32);

  __shared__ float red[4];
  if ((t & 31) == 0) red[t >> 5] = part;
  __syncthreads();
  const float ew = red[0] + red[1] + red[2] + red[3];

  float4 vs4 = ((const float4*)(kvc + D_))[t];
  float4 vv4 = ((const float4*)(kvc + 2 * D_))[t];
  float4 vc4 = ((const float4*)(v + (size_t)col * D_))[t];

  float* as = agg_s + (size_t)row * D_ + t * 4;
  float* av = agg_v + (size_t)row * D_ + t * 4;
  atomicAdd(as + 0, ew * vs4.x);
  atomicAdd(as + 1, ew * vs4.y);
  atomicAdd(as + 2, ew * vs4.z);
  atomicAdd(as + 3, ew * vs4.w);
  atomicAdd(av + 0, ew * vv4.x * vc4.x);
  atomicAdd(av + 1, ew * vv4.y * vc4.y);
  atomicAdd(av + 2, ew * vv4.z * vc4.z);
  atomicAdd(av + 3, ew * vv4.w * vc4.w);
}

// ---------------------------------------------------------------------------
// x += agg_s (also emit bf16 x); agg_v -> bf16 for the Wvm GEMM
// ---------------------------------------------------------------------------
__global__ void update_x_kernel(float* __restrict__ x,
                                const float* __restrict__ agg_s,
                                const float* __restrict__ agg_v,
                                unsigned short* __restrict__ xbf,
                                unsigned short* __restrict__ avbf) {
  int i = blockIdx.x * blockDim.x + threadIdx.x;   // over BN*D
  float nx = x[i] + agg_s[i];
  x[i] = nx;
  xbf[i] = f2bf(nx);
  avbf[i] = f2bf(agg_v[i]);
}

// ---------------------------------------------------------------------------
// out = where(mask, 0, x); also write mask floats (second output of the tuple)
// ---------------------------------------------------------------------------
__global__ void final_kernel(const float* __restrict__ x,
                             const int* __restrict__ tok,
                             float* __restrict__ out,
                             float* __restrict__ mask_out) {
  int i  = blockIdx.x * blockDim.x + threadIdx.x;  // over BN*D
  int bn = i / D_;
  bool pm = (tok[bn] == PAD_);
  out[i] = pm ? 0.0f : x[i];
  if ((i & (D_ - 1)) == 0) mask_out[bn] = pm ? 1.0f : 0.0f;
}

// ---------------------------------------------------------------------------
// Host driver
// ---------------------------------------------------------------------------
extern "C" void kernel_launch(void* const* d_in, const int* in_sizes, int n_in,
                              void* d_out, int out_size, void* d_ws, size_t ws_size,
                              hipStream_t stream) {
  (void)in_sizes; (void)n_in; (void)out_size; (void)ws_size;

  const int*   src_tokens = (const int*)d_in[0];
  const float* coords     = (const float*)d_in[1];
  // d_in[2] (src_distance) and d_in[3] (src_edge_type) are unused by reference
  const int*   edge_index = (const int*)d_in[4];
  const float* embed      = (const float*)d_in[5];
  const float* vw         = (const float*)d_in[6];
  const float* vb         = (const float*)d_in[7];
  const float* Wq         = (const float*)d_in[8];
  const float* bq         = (const float*)d_in[9];
  const float* Wkv        = (const float*)d_in[10];
  const float* bkv        = (const float*)d_in[11];
  const float* Wvm        = (const float*)d_in[12];
  const float* bvm        = (const float*)d_in[13];

  float* out      = (float*)d_out;                 // [BN*D] then [BN] mask
  float* mask_out = out + (size_t)BN_ * D_;

  // ---- workspace carve-out (~161 MB total) ----
  char* p = (char*)d_ws;
  auto carve = [&](size_t bytes) {
    char* r = p;
    p += (bytes + 255) & ~(size_t)255;
    return (void*)r;
  };
  float* x    = (float*)carve((size_t)BN_ * D_ * 4);
  float* v    = (float*)carve((size_t)BN_ * D_ * 4);
  float* q    = (float*)carve((size_t)BN_ * D_ * 4);
  float* kv   = (float*)carve((size_t)BN_ * 3 * D_ * 4);
  float* aggs = (float*)carve((size_t)BN_ * D_ * 4);
  float* aggv = (float*)carve((size_t)BN_ * D_ * 4);
  unsigned short* xbf   = (unsigned short*)carve((size_t)BN_ * D_ * 2);
  unsigned short* avbf  = (unsigned short*)carve((size_t)BN_ * D_ * 2);
  unsigned short* WqBt  = (unsigned short*)carve((size_t)L_ * D_ * D_ * 2);
  unsigned short* WkvBt = (unsigned short*)carve((size_t)L_ * D_ * 3 * D_ * 2);
  unsigned short* WvmBt = (unsigned short*)carve((size_t)L_ * D_ * D_ * 2);

  // ---- weights -> bf16, transposed to [Nc][K] once per launch ----
  {
    int n1 = L_ * D_ * D_;        // 1,048,576
    int n2 = L_ * D_ * 3 * D_;    // 3,145,728
    f2bf_transpose_kernel<<<n1 / 256, 256, 0, stream>>>(Wq,  WqBt,  D_, D_);
    f2bf_transpose_kernel<<<n2 / 256, 256, 0, stream>>>(Wkv, WkvBt, D_, 3 * D_);
    f2bf_transpose_kernel<<<n1 / 256, 256, 0, stream>>>(Wvm, WvmBt, D_, D_);
  }

  // ---- embedding + vector projection ----
  embed_kernel<<<(BN_ * D_ / 4) / 256, 256, 0, stream>>>(src_tokens, embed, x, xbf);
  vecproj_kernel<<<(BN_ * D_) / 256, 256, 0, stream>>>(coords, vw, vb, v);

  const dim3 gemm_blk(256);
  for (int l = 0; l < L_; ++l) {
    // q = x @ Wq[l] + bq[l]
    gemm_bf16_kernel<<<dim3(D_ / TN, BN_ / TM), gemm_blk, 0, stream>>>(
        xbf, WqBt + (size_t)l * D_ * D_, bq + (size_t)l * D_,
        nullptr, q, BN_, D_, D_);
    // kv = x @ Wkv[l] + bkv[l]
    gemm_bf16_kernel<<<dim3(3 * D_ / TN, BN_ / TM), gemm_blk, 0, stream>>>(
        xbf, WkvBt + (size_t)l * D_ * 3 * D_, bkv + (size_t)l * 3 * D_,
        nullptr, kv, BN_, 3 * D_, D_);

    (void)hipMemsetAsync(aggs, 0, (size_t)BN_ * D_ * 4, stream);
    (void)hipMemsetAsync(aggv, 0, (size_t)BN_ * D_ * 4, stream);

    edge_kernel<<<E_, 128, 0, stream>>>(edge_index, q, kv, v, aggs, aggv);

    update_x_kernel<<<(BN_ * D_) / 256, 256, 0, stream>>>(x, aggs, aggv, xbf, avbf);

    // v = v + agg_v @ Wvm[l] + bvm[l]   (fused C-add, in place)
    gemm_bf16_kernel<<<dim3(D_ / TN, BN_ / TM), gemm_blk, 0, stream>>>(
        avbf, WvmBt + (size_t)l * D_ * D_, bvm + (size_t)l * D_,
        v, v, BN_, D_, D_);
  }

  final_kernel<<<(BN_ * D_) / 256, 256, 0, stream>>>(x, src_tokens, out, mask_out);
}
